// PairformerLayer_20306605375685
// MI455X (gfx1250) — compile-verified, matrix operand order checked by reading, not run on hardware
//
#include <hip/hip_runtime.h>
#include <math.h>

// ---------------------------------------------------------------------------
// Pairformer layer for MI455X (gfx1250, wave32, WMMA).
// All matmuls run through v_wmma_f32_16x16x32_bf16 (bf16 in, f32 accum).
// Sizes: B=1, N=192, TS=384, TZ=128, H_S=16(d=24), H_P=4(c=32).
// d_out = [s (192*384) | z (192*192*128)] fp32, used as the residual streams.
// Workspace requirement: ~177 MB.
// GEMMs are double-buffered: global loads for tile k+1 issue before the WMMA
// on tile k so the s_wait_loadcnt lands after the matrix op.
// ---------------------------------------------------------------------------

#define NRES 192
#define NSQ  36864           // 192*192
#define TZC  128
#define TSC  384

typedef __attribute__((ext_vector_type(16))) __bf16 v16bf_t;
typedef __attribute__((ext_vector_type(8)))  float  v8f_t;
typedef __attribute__((ext_vector_type(4)))  __bf16 v4bf_t;
typedef __attribute__((ext_vector_type(2)))  __bf16 v2bf_t;

__device__ __forceinline__ v8f_t wmma_bf16(v16bf_t a, v16bf_t b, v8f_t c) {
  return __builtin_amdgcn_wmma_f32_16x16x32_bf16(false, a, false, b, (short)0, c,
                                                 false, false);
}

// A/B fragment gather from an LDS row (row-major, contraction contiguous).
// Per ISA 7.12.2 (16-bit A 16x32): lane half selects K offset 0/8 (low VGPRs)
// and 16/24 (high VGPRs); each VGPR packs two consecutive K values.
__device__ __forceinline__ v16bf_t load_frag_row(const __bf16* base, int half) {
  v16bf_t f{};
#pragma unroll
  for (int v = 0; v < 8; ++v) {
    int k0 = ((v < 4) ? 2 * v : 16 + 2 * (v - 4)) + 8 * half;
    f[2 * v]     = base[k0];
    f[2 * v + 1] = base[k0 + 1];
  }
  return f;
}

__device__ __forceinline__ v4bf_t pack4(float a, float b, float c, float d) {
  v4bf_t r;
  r[0] = (__bf16)a; r[1] = (__bf16)b; r[2] = (__bf16)c; r[3] = (__bf16)d;
  return r;
}

__device__ __forceinline__ float sigmoidf_(float x) { return 1.f / (1.f + expf(-x)); }

// ---------------------------------------------------------------------------
// Generic GEMM: C[M,N] = act(A[M,K] @ B[K,N] + bias)
// act: 0 = none, 1 = sigmoid, 2 = silu.
// Requirements: M % 64 == 0, K % 32 == 0, A rows 16B-aligned. N arbitrary.
// Tile 64(M) x 32(N), 8 waves -> 4x2 grid of 16x16 WMMA subtiles.
// Double-buffered LDS: loads for tile k+1 overlap the WMMA on tile k.
// ---------------------------------------------------------------------------
__global__ __launch_bounds__(256)
void gemm_bf16_kernel(const float* __restrict__ A, const float* __restrict__ B,
                      const float* __restrict__ bias, float* __restrict__ C,
                      int M, int K, int Nn, int act) {
  __shared__ __bf16 As[2][64][32];
  __shared__ __bf16 Bs[2][32][32];   // stored transposed: [n][k]
  const int tid = threadIdx.x, lane = tid & 31, w = tid >> 5;
  const int wm = w >> 1, wn = w & 1;
  const int half = lane >> 4, lr = lane & 15;
  const int m0 = blockIdx.y * 64, n0 = blockIdx.x * 32;
  // A staging: 2 float4 per thread (64x32 = 512 float4)
  const int ar0 = (tid * 2) >> 3, ac0 = ((tid * 2) & 7) * 4;
  const int ar1 = (tid * 2 + 1) >> 3, ac1 = ((tid * 2 + 1) & 7) * 4;
  const float* arow0 = A + (size_t)(m0 + ar0) * K + ac0;
  const float* arow1 = A + (size_t)(m0 + ar1) * K + ac1;
  // B staging: thread -> (n, 4 consecutive k), strided global reads
  const int bn = tid & 31, bkg = (tid >> 5) * 4;
  const bool bvalid = (n0 + bn) < Nn;
  const float* bcol = B + (size_t)bkg * Nn + (n0 + bn);
  const int ntiles = K >> 5;

  // prologue: stage tile 0
  {
    const float4 a0 = *(const float4*)(arow0);
    const float4 a1 = *(const float4*)(arow1);
    *(v4bf_t*)&As[0][ar0][ac0] = pack4(a0.x, a0.y, a0.z, a0.w);
    *(v4bf_t*)&As[0][ar1][ac1] = pack4(a1.x, a1.y, a1.z, a1.w);
    float b0 = 0.f, b1 = 0.f, b2 = 0.f, b3 = 0.f;
    if (bvalid) { b0 = bcol[0]; b1 = bcol[Nn]; b2 = bcol[2 * (size_t)Nn]; b3 = bcol[3 * (size_t)Nn]; }
    *(v4bf_t*)&Bs[0][bn][bkg] = pack4(b0, b1, b2, b3);
  }
  __syncthreads();

  v8f_t acc = {};
  for (int kt = 0; kt < ntiles; ++kt) {
    const int cur = kt & 1, nxt = cur ^ 1;
    const bool have_next = (kt + 1 < ntiles);
    float4 na0 = {}, na1 = {};
    float nb0 = 0.f, nb1 = 0.f, nb2 = 0.f, nb3 = 0.f;
    if (have_next) {
      const int k0 = (kt + 1) << 5;
      na0 = *(const float4*)(arow0 + k0);
      na1 = *(const float4*)(arow1 + k0);
      if (bvalid) {
        const float* bp = bcol + (size_t)k0 * Nn;
        nb0 = bp[0]; nb1 = bp[Nn]; nb2 = bp[2 * (size_t)Nn]; nb3 = bp[3 * (size_t)Nn];
      }
      if (kt + 2 < ntiles)
        __builtin_prefetch(arow0 + ((kt + 2) << 5), 0, 1);
    }
    v16bf_t af = load_frag_row(&As[cur][wm * 16 + lr][0], half);
    v16bf_t bf = load_frag_row(&Bs[cur][wn * 16 + lr][0], half);
    acc = wmma_bf16(af, bf, acc);
    __syncthreads();
    if (have_next) {
      *(v4bf_t*)&As[nxt][ar0][ac0] = pack4(na0.x, na0.y, na0.z, na0.w);
      *(v4bf_t*)&As[nxt][ar1][ac1] = pack4(na1.x, na1.y, na1.z, na1.w);
      *(v4bf_t*)&Bs[nxt][bn][bkg]  = pack4(nb0, nb1, nb2, nb3);
    }
    __syncthreads();
  }
  int gn = n0 + wn * 16 + lr;
  if (gn < Nn) {
    float bv = bias ? bias[gn] : 0.f;
#pragma unroll
    for (int v = 0; v < 8; ++v) {
      int gm = m0 + wm * 16 + v + 8 * half;
      float x = acc[v] + bv;
      if (act == 1)      x = sigmoidf_(x);
      else if (act == 2) x = x * sigmoidf_(x);
      C[(size_t)gm * Nn + gn] = x;
    }
  }
}

// ---------------------------------------------------------------------------
// Triangle-multiply batched GEMM: for each channel d (=blockIdx.z),
// out[i,j,d] = sum_k A[d][i][k] * B[d][j][k]   (A,B pre-packed bf16 192x192)
// Double-buffered like the generic GEMM.
// ---------------------------------------------------------------------------
__global__ __launch_bounds__(256)
void trimul_bmm_kernel(const __bf16* __restrict__ A, const __bf16* __restrict__ Bm,
                       float* __restrict__ out) {
  const int d = blockIdx.z;
  const int m0 = blockIdx.y * 64, n0 = blockIdx.x * 32;
  __shared__ __bf16 As[2][64][32];
  __shared__ __bf16 Bs[2][32][32];
  const int tid = threadIdx.x, lane = tid & 31, w = tid >> 5;
  const int wm = w >> 1, wn = w & 1, half = lane >> 4, lr = lane & 15;
  const __bf16* Ad = A + (size_t)d * NSQ;
  const __bf16* Bd = Bm + (size_t)d * NSQ;
  const int arr = tid >> 2, ac8 = (tid & 3) * 8;     // 16B per thread (A)
  const int bnn = tid >> 3, bk4 = (tid & 7) * 4;     // 8B per thread (B)
  const __bf16* arow = Ad + (size_t)(m0 + arr) * 192 + ac8;
  const __bf16* brow = Bd + (size_t)(n0 + bnn) * 192 + bk4;

  *(float4*)&As[0][arr][ac8] = *(const float4*)(arow);
  *(float2*)&Bs[0][bnn][bk4] = *(const float2*)(brow);
  __syncthreads();

  v8f_t acc = {};
  for (int kt = 0; kt < 6; ++kt) {
    const int cur = kt & 1, nxt = cur ^ 1;
    const bool have_next = (kt < 5);
    float4 na = {};
    float2 nb = {};
    if (have_next) {
      const int k0 = (kt + 1) << 5;
      na = *(const float4*)(arow + k0);
      nb = *(const float2*)(brow + k0);
      if (kt < 4)
        __builtin_prefetch(arow + ((kt + 2) << 5), 0, 1);
    }
    v16bf_t af = load_frag_row(&As[cur][wm * 16 + lr][0], half);
    v16bf_t bf = load_frag_row(&Bs[cur][wn * 16 + lr][0], half);
    acc = wmma_bf16(af, bf, acc);
    __syncthreads();
    if (have_next) {
      *(float4*)&As[nxt][arr][ac8] = na;
      *(float2*)&Bs[nxt][bnn][bk4] = nb;
    }
    __syncthreads();
  }
  int j = n0 + wn * 16 + lr;
#pragma unroll
  for (int v = 0; v < 8; ++v) {
    int i = m0 + wm * 16 + v + 8 * half;
    out[((size_t)i * 192 + j) * TZC + d] = acc[v];
  }
}

// ---------------------------------------------------------------------------
// LayerNorm over last dim D (rows = gridDim.x), eps 1e-5.
// ---------------------------------------------------------------------------
__global__ __launch_bounds__(128)
void ln_kernel(const float* __restrict__ in, const float* __restrict__ g,
               const float* __restrict__ b, float* __restrict__ out, int D) {
  __shared__ float red[128];
  const size_t row = blockIdx.x;
  const float* x = in + row * D;
  float* o = out + row * D;
  const int tid = threadIdx.x;
  float s = 0.f;
  for (int c = tid; c < D; c += 128) s += x[c];
  red[tid] = s; __syncthreads();
  for (int st = 64; st > 0; st >>= 1) { if (tid < st) red[tid] += red[tid + st]; __syncthreads(); }
  float mean = red[0] / D; __syncthreads();
  float vs = 0.f;
  for (int c = tid; c < D; c += 128) { float t = x[c] - mean; vs += t * t; }
  red[tid] = vs; __syncthreads();
  for (int st = 64; st > 0; st >>= 1) { if (tid < st) red[tid] += red[tid + st]; __syncthreads(); }
  float rstd = rsqrtf(red[0] / D + 1e-5f);
  for (int c = tid; c < D; c += 128) o[c] = (x[c] - mean) * rstd * g[c] + b[c];
}

// ---------------------------------------------------------------------------
// Elementwise kernels
// ---------------------------------------------------------------------------
__global__ void ew_pg_mask_kernel(float* __restrict__ p, const float* __restrict__ g,
                                  const float* __restrict__ pm, size_t n) {
  size_t i = (size_t)blockIdx.x * 256 + threadIdx.x;
  if (i < n) p[i] *= g[i] * pm[i >> 8];          // 256 channels per pair row
}
__global__ void ew_mul_kernel(float* __restrict__ d, const float* __restrict__ a,
                              const float* __restrict__ b, size_t n) {
  size_t i = (size_t)blockIdx.x * 256 + threadIdx.x;
  if (i < n) d[i] = a[i] * b[i];
}
__global__ void ew_fma_add_kernel(float* __restrict__ d, const float* __restrict__ a,
                                  const float* __restrict__ b, size_t n) {
  size_t i = (size_t)blockIdx.x * 256 + threadIdx.x;
  if (i < n) d[i] += a[i] * b[i];
}
__global__ void ew_add_kernel(float* __restrict__ d, const float* __restrict__ a, size_t n) {
  size_t i = (size_t)blockIdx.x * 256 + threadIdx.x;
  if (i < n) d[i] += a[i];
}
__global__ void ew_add_tr128_kernel(float* __restrict__ d, const float* __restrict__ a) {
  size_t idx = (size_t)blockIdx.x * 256 + threadIdx.x;   // < NSQ*128
  int c = idx & 127;
  size_t r = idx >> 7;
  int j = (int)(r % 192), i = (int)(r / 192);
  d[idx] += a[(((size_t)j * 192) + i) * TZC + c];
}
__global__ void tr128_kernel(float* __restrict__ o, const float* __restrict__ in) {
  size_t idx = (size_t)blockIdx.x * 256 + threadIdx.x;   // < NSQ*128
  int c = idx & 127;
  size_t r = idx >> 7;
  int j = (int)(r % 192), i = (int)(r / 192);
  o[idx] = in[(((size_t)j * 192) + i) * TZC + c];
}
__global__ void trmask_kernel(float* __restrict__ o, const float* __restrict__ in) {
  int idx = blockIdx.x * 256 + threadIdx.x;              // < NSQ
  int j = idx % 192, i = idx / 192;
  o[idx] = in[j * 192 + i];
}

// Repack gated pg = [a|b] (NSQ x 256 f32, row = (i,k)) into per-channel bf16
// matrices At[d][i][k], Bt[d][j][k].  outgoing: a[i,k,d]; incoming: a[k,i,d].
// Each thread handles two consecutive k so stores are 4-byte.
__global__ void pack_ab_kernel(const float* __restrict__ pg, __bf16* __restrict__ At,
                               __bf16* __restrict__ Bt, int outgoing) {
  size_t idx = (size_t)blockIdx.x * 256 + threadIdx.x;   // < 128*192*96
  if (idx >= (size_t)TZC * NSQ / 2) return;
  int kp = (int)(idx % 96) * 2;
  int d  = (int)((idx / 96) % TZC);
  int i  = (int)(idx / (96 * TZC));
  size_t s0, s1;
  if (outgoing) { s0 = ((size_t)i * 192 + kp) * 256 + d; s1 = s0 + 256; }
  else          { s0 = ((size_t)kp * 192 + i) * 256 + d; s1 = s0 + (size_t)192 * 256; }
  size_t dst = ((size_t)d * 192 + i) * 192 + kp;
  v2bf_t a; a[0] = (__bf16)pg[s0];       a[1] = (__bf16)pg[s1];
  v2bf_t b; b[0] = (__bf16)pg[s0 + 128]; b[1] = (__bf16)pg[s1 + 128];
  *(v2bf_t*)(At + dst) = a;
  *(v2bf_t*)(Bt + dst) = b;
}

// ---------------------------------------------------------------------------
// Fused triangle attention: one block per (qtile, i, h). C_P = 32 == WMMA K.
// q unscaled; scale applied to scores. q/k/v layout [(i*192+n)*128+h*32+c].
// tb layout [(q*192+k)*4+h]; pm row i gives key mask.
// ---------------------------------------------------------------------------
__global__ __launch_bounds__(256)
void tri_att_kernel(const float* __restrict__ q, const float* __restrict__ k,
                    const float* __restrict__ v, const float* __restrict__ tb,
                    const float* __restrict__ pm, float* __restrict__ o, float scale) {
  const int qt = blockIdx.x, i = blockIdx.y, h = blockIdx.z;
  __shared__ __bf16 Qs[32][32];
  __shared__ __bf16 Ks[32][32];
  __shared__ __bf16 Vs[192][32];
  __shared__ __bf16 Ps[32][192];
  __shared__ float  Sc[32][192];
  const int tid = threadIdx.x, lane = tid & 31, w = tid >> 5;
  const int half = lane >> 4, lr = lane & 15, wm = w >> 1, wn = w & 1;
  const int q0 = qt * 32;
  const size_t rowbase = (size_t)i * 192;
  const int r32 = tid >> 3, c4 = (tid & 7) * 4;     // 32x32 staging: 1 float4/thread
  {
    const float4 a = *(const float4*)(q + (rowbase + q0 + r32) * TZC + h * 32 + c4);
    *(v4bf_t*)&Qs[r32][c4] = pack4(a.x, a.y, a.z, a.w);
  }
#pragma unroll
  for (int t = 0; t < 6; ++t) {                      // V: 192x32 = 1536 float4
    int vidx = tid + 256 * t;
    int r = vidx >> 3, cc = (vidx & 7) * 4;
    const float4 a = *(const float4*)(v + (rowbase + r) * TZC + h * 32 + cc);
    *(v4bf_t*)&Vs[r][cc] = pack4(a.x, a.y, a.z, a.w);
  }
  for (int kt = 0; kt < 6; ++kt) {
    {
      const float4 a =
          *(const float4*)(k + (rowbase + kt * 32 + r32) * TZC + h * 32 + c4);
      *(v4bf_t*)&Ks[r32][c4] = pack4(a.x, a.y, a.z, a.w);
    }
    __syncthreads();
    if (w < 4) {
      v8f_t acc = {};
      v16bf_t af = load_frag_row(&Qs[wm * 16 + lr][0], half);
      v16bf_t bf = load_frag_row(&Ks[wn * 16 + lr][0], half);
      acc = wmma_bf16(af, bf, acc);
      int col = kt * 32 + wn * 16 + lr;
#pragma unroll
      for (int vv = 0; vv < 8; ++vv) Sc[wm * 16 + vv + 8 * half][col] = acc[vv];
    }
    __syncthreads();
  }
  if (tid < 32) {
    int r = tid, qg = q0 + r;
    float mx = -1e30f;
    for (int j = 0; j < 192; ++j) {
      float sc = Sc[r][j] * scale + tb[((size_t)qg * 192 + j) * 4 + h]
                 + 1e9f * (pm[rowbase + j] - 1.f);
      Sc[r][j] = sc; mx = fmaxf(mx, sc);
    }
    float sum = 0.f;
    for (int j = 0; j < 192; ++j) { float e = expf(Sc[r][j] - mx); sum += e; Sc[r][j] = e; }
    float inv = 1.f / sum;
    for (int j = 0; j < 192; j += 2) {
      v2bf_t p; p[0] = (__bf16)(Sc[r][j] * inv); p[1] = (__bf16)(Sc[r][j + 1] * inv);
      *(v2bf_t*)&Ps[r][j] = p;
    }
  }
  __syncthreads();
  if (w < 4) {
    v8f_t acc = {};
    for (int kt = 0; kt < 6; ++kt) {
      v16bf_t af{}, bf{};
      const __bf16* prow = &Ps[wm * 16 + lr][kt * 32];
#pragma unroll
      for (int vv = 0; vv < 8; ++vv) {
        int kk = ((vv < 4) ? 2 * vv : 16 + 2 * (vv - 4)) + 8 * half;
        af[2 * vv]     = prow[kk];
        af[2 * vv + 1] = prow[kk + 1];
        bf[2 * vv]     = Vs[kt * 32 + kk][wn * 16 + lr];
        bf[2 * vv + 1] = Vs[kt * 32 + kk + 1][wn * 16 + lr];
      }
      acc = wmma_bf16(af, bf, acc);
    }
    int c = wn * 16 + lr;
#pragma unroll
    for (int vv = 0; vv < 8; ++vv) {
      int qg = q0 + wm * 16 + vv + 8 * half;
      o[(rowbase + qg) * TZC + h * 32 + c] = acc[vv];
    }
  }
}

// ---------------------------------------------------------------------------
// Fused single-track attention with pair bias: block per (qtile, h).
// Head dim 24 zero-padded to 32 (pad is whole float4 groups -> uniform branch).
// zb layout [(i*192+j)*16+h]. msk over keys.
// ---------------------------------------------------------------------------
__global__ __launch_bounds__(256)
void s_att_kernel(const float* __restrict__ q, const float* __restrict__ k,
                  const float* __restrict__ v, const float* __restrict__ zb,
                  const float* __restrict__ msk, float* __restrict__ o, float scale) {
  const int qt = blockIdx.x, h = blockIdx.y;
  __shared__ __bf16 Qs[32][32];
  __shared__ __bf16 Ks[32][32];
  __shared__ __bf16 Vs[192][32];
  __shared__ __bf16 Ps[32][192];
  __shared__ float  Sc[32][192];
  const int tid = threadIdx.x, lane = tid & 31, w = tid >> 5;
  const int half = lane >> 4, lr = lane & 15, wm = w >> 1, wn = w & 1;
  const int q0 = qt * 32;
  const int r32 = tid >> 3, c4 = (tid & 7) * 4;
  const v4bf_t z4 = pack4(0.f, 0.f, 0.f, 0.f);
  if (c4 < 24) {
    const float4 a = *(const float4*)(q + (size_t)(q0 + r32) * TSC + h * 24 + c4);
    *(v4bf_t*)&Qs[r32][c4] = pack4(a.x, a.y, a.z, a.w);
  } else {
    *(v4bf_t*)&Qs[r32][c4] = z4;
  }
#pragma unroll
  for (int t = 0; t < 6; ++t) {
    int vidx = tid + 256 * t;
    int r = vidx >> 3, cc = (vidx & 7) * 4;
    if (cc < 24) {
      const float4 a = *(const float4*)(v + (size_t)r * TSC + h * 24 + cc);
      *(v4bf_t*)&Vs[r][cc] = pack4(a.x, a.y, a.z, a.w);
    } else {
      *(v4bf_t*)&Vs[r][cc] = z4;
    }
  }
  for (int kt = 0; kt < 6; ++kt) {
    if (c4 < 24) {
      const float4 a =
          *(const float4*)(k + (size_t)(kt * 32 + r32) * TSC + h * 24 + c4);
      *(v4bf_t*)&Ks[r32][c4] = pack4(a.x, a.y, a.z, a.w);
    } else {
      *(v4bf_t*)&Ks[r32][c4] = z4;
    }
    __syncthreads();
    if (w < 4) {
      v8f_t acc = {};
      v16bf_t af = load_frag_row(&Qs[wm * 16 + lr][0], half);
      v16bf_t bf = load_frag_row(&Ks[wn * 16 + lr][0], half);
      acc = wmma_bf16(af, bf, acc);
      int col = kt * 32 + wn * 16 + lr;
#pragma unroll
      for (int vv = 0; vv < 8; ++vv) Sc[wm * 16 + vv + 8 * half][col] = acc[vv];
    }
    __syncthreads();
  }
  if (tid < 32) {
    int r = tid, qg = q0 + r;
    float mx = -1e30f;
    for (int j = 0; j < 192; ++j) {
      float sc = Sc[r][j] * scale + zb[((size_t)qg * 192 + j) * 16 + h]
                 + 1e6f * (msk[j] - 1.f);
      Sc[r][j] = sc; mx = fmaxf(mx, sc);
    }
    float sum = 0.f;
    for (int j = 0; j < 192; ++j) { float e = expf(Sc[r][j] - mx); sum += e; Sc[r][j] = e; }
    float inv = 1.f / sum;
    for (int j = 0; j < 192; j += 2) {
      v2bf_t p; p[0] = (__bf16)(Sc[r][j] * inv); p[1] = (__bf16)(Sc[r][j + 1] * inv);
      *(v2bf_t*)&Ps[r][j] = p;
    }
  }
  __syncthreads();
  if (w < 4) {
    v8f_t acc = {};
    for (int kt = 0; kt < 6; ++kt) {
      v16bf_t af{}, bf{};
      const __bf16* prow = &Ps[wm * 16 + lr][kt * 32];
#pragma unroll
      for (int vv = 0; vv < 8; ++vv) {
        int kk = ((vv < 4) ? 2 * vv : 16 + 2 * (vv - 4)) + 8 * half;
        af[2 * vv]     = prow[kk];
        af[2 * vv + 1] = prow[kk + 1];
        bf[2 * vv]     = Vs[kt * 32 + kk][wn * 16 + lr];
        bf[2 * vv + 1] = Vs[kt * 32 + kk + 1][wn * 16 + lr];
      }
      acc = wmma_bf16(af, bf, acc);
    }
    int c = wn * 16 + lr;
    if (c < 24) {
#pragma unroll
      for (int vv = 0; vv < 8; ++vv) {
        int qg = q0 + wm * 16 + vv + 8 * half;
        o[(size_t)qg * TSC + h * 24 + c] = acc[vv];
      }
    }
  }
}

// ---------------------------------------------------------------------------
// Host orchestration
// ---------------------------------------------------------------------------
// Flattened input indices (setup_inputs dict order, nested dicts in order):
enum {
  IN_S = 0, IN_Z = 1, IN_MASK = 2, IN_PMASK = 3,
  P_PRE_G = 4, P_PRE_B = 5,
  A_WQ = 6, A_BQ = 7, A_WK = 8, A_WV = 9, A_WG = 10,
  A_ZLN_G = 11, A_ZLN_B = 12, A_WZ = 13, A_WO = 14,
  TMO = 15,   // tri_mul_out: +0 ng,+1 nb,+2 p_in,+3 g_in,+4 og,+5 ob,+6 p_out,+7 g_out
  TMI = 23,
  TAS = 31,   // tri_att: +0 lg,+1 lb,+2 tri_w,+3 wq,+4 wk,+5 wv,+6 wg,+7 bg,+8 wo,+9 bo
  TAE = 41,
  TRS = 51,   // transition: +0 lg,+1 lb,+2 w1,+3 w2,+4 w3
  TRZ = 56,
};

extern "C" void kernel_launch(void* const* d_in, const int* in_sizes, int n_in,
                              void* d_out, int out_size, void* d_ws, size_t ws_size,
                              hipStream_t stream) {
  (void)in_sizes; (void)n_in; (void)out_size; (void)ws_size;
  auto P = [&](int i) -> const float* { return (const float*)d_in[i]; };

  float* sr = (float*)d_out;                 // s residual stream (192*384)
  float* zr = sr + (size_t)NRES * TSC;       // z residual stream (NSQ*128)
  hipMemcpyAsync(sr, d_in[IN_S], (size_t)NRES * TSC * sizeof(float),
                 hipMemcpyDeviceToDevice, stream);
  hipMemcpyAsync(zr, d_in[IN_Z], (size_t)NSQ * TZC * sizeof(float),
                 hipMemcpyDeviceToDevice, stream);

  float* ws = (float*)d_ws;
  size_t off = 0;
  auto alloc = [&](size_t n) { float* p = ws + off; off += n; return p; };
  float* BUF0 = alloc((size_t)NSQ * 256);
  float* BUF1 = alloc((size_t)NSQ * 256);
  float* BUF2 = alloc((size_t)NSQ * 128);
  float* BUF3 = alloc((size_t)NSQ * 128);
  float* BUF4 = alloc((size_t)NSQ * 128);
  float* BUF5 = alloc((size_t)NSQ * 128);
  float* TBUF = alloc((size_t)NSQ * 16);     // tri bias (N=4) / zb (N=16)
  float* PMT  = alloc((size_t)NSQ);          // transposed pair mask
  __bf16* At  = (__bf16*)alloc((size_t)NSQ * 64);  // NSQ*128 bf16
  __bf16* Bt  = (__bf16*)alloc((size_t)NSQ * 64);
  float* SN  = alloc((size_t)NRES * TSC);
  float* SQ  = alloc((size_t)NRES * TSC);
  float* SK  = alloc((size_t)NRES * TSC);
  float* SV  = alloc((size_t)NRES * TSC);
  float* SG  = alloc((size_t)NRES * TSC);
  float* SO  = alloc((size_t)NRES * TSC);
  float* SH1 = alloc((size_t)NRES * 1536);
  float* SH2 = alloc((size_t)NRES * 1536);

  auto gemm = [&](const float* A, const float* B, const float* bias, float* C,
                  int M, int K, int Nn, int act) {
    dim3 g((Nn + 31) / 32, (M + 63) / 64);
    gemm_bf16_kernel<<<g, 256, 0, stream>>>(A, B, bias, C, M, K, Nn, act);
  };
  auto ln = [&](const float* x, const float* g, const float* b, float* o,
                int rows, int D) {
    ln_kernel<<<rows, 128, 0, stream>>>(x, g, b, o, D);
  };
  auto ewmul = [&](float* d, const float* a, const float* b, size_t n) {
    ew_mul_kernel<<<(unsigned)((n + 255) / 256), 256, 0, stream>>>(d, a, b, n);
  };
  auto ewfma = [&](float* d, const float* a, const float* b, size_t n) {
    ew_fma_add_kernel<<<(unsigned)((n + 255) / 256), 256, 0, stream>>>(d, a, b, n);
  };
  auto ewadd = [&](float* d, const float* a, size_t n) {
    ew_add_kernel<<<(unsigned)((n + 255) / 256), 256, 0, stream>>>(d, a, n);
  };

  // ---- TriangleMultiplication (outgoing then incoming) -------------------
  auto tri_mul = [&](int base, int outgoing) {
    ln(zr, P(base + 0), P(base + 1), BUF2, NSQ, 128);                 // x
    gemm(BUF2, P(base + 2), nullptr, BUF0, NSQ, 128, 256, 0);         // p_in
    gemm(BUF2, P(base + 3), nullptr, BUF1, NSQ, 128, 256, 1);         // sigm(g_in)
    ew_pg_mask_kernel<<<NSQ, 256, 0, stream>>>(BUF0, BUF1, P(IN_PMASK),
                                               (size_t)NSQ * 256);
    pack_ab_kernel<<<(unsigned)(((size_t)NSQ * 64) / 256), 256, 0, stream>>>(
        BUF0, At, Bt, outgoing);
    trimul_bmm_kernel<<<dim3(6, 3, 128), 256, 0, stream>>>(At, Bt, BUF3);
    ln(BUF3, P(base + 4), P(base + 5), BUF4, NSQ, 128);               // LN(t)
    gemm(BUF4, P(base + 6), nullptr, BUF5, NSQ, 128, 128, 0);         // p_out
    gemm(BUF2, P(base + 7), nullptr, BUF0, NSQ, 128, 128, 1);         // sigm(g_out)
    ewfma(zr, BUF5, BUF0, (size_t)NSQ * 128);                         // z += proj*gate
  };
  tri_mul(TMO, 1);
  tri_mul(TMI, 0);

  // ---- TriangleAttention (starting then ending) --------------------------
  auto tri_att = [&](const float* zsrc, const float* pmask, int base, bool trans_out) {
    ln(zsrc, P(base + 0), P(base + 1), BUF2, NSQ, 128);               // x
    gemm(BUF2, P(base + 2), nullptr, TBUF, NSQ, 128, 4, 0);           // tri bias
    gemm(BUF2, P(base + 3), nullptr, BUF0, NSQ, 128, 128, 0);         // q
    gemm(BUF2, P(base + 4), nullptr, BUF1, NSQ, 128, 128, 0);         // k
    gemm(BUF2, P(base + 5), nullptr, BUF3, NSQ, 128, 128, 0);         // v
    gemm(BUF2, P(base + 6), P(base + 7), BUF4, NSQ, 128, 128, 1);     // sigm(g+bg)
    tri_att_kernel<<<dim3(6, 192, 4), 256, 0, stream>>>(
        BUF0, BUF1, BUF3, TBUF, pmask, BUF5, 0.17677669529663689f);   // 1/sqrt(32)
    ewmul(BUF0, BUF4, BUF5, (size_t)NSQ * 128);                       // g*o
    gemm(BUF0, P(base + 8), P(base + 9), BUF1, NSQ, 128, 128, 0);     // @wo + bo
    if (trans_out)
      ew_add_tr128_kernel<<<(unsigned)(((size_t)NSQ * 128) / 256), 256, 0,
                            stream>>>(zr, BUF1);
    else
      ewadd(zr, BUF1, (size_t)NSQ * 128);
  };
  tri_att(zr, P(IN_PMASK), TAS, false);
  // ending node: run on z^T with mask^T, add transposed output
  tr128_kernel<<<(unsigned)(((size_t)NSQ * 128) / 256), 256, 0, stream>>>(BUF0, zr);
  trmask_kernel<<<NSQ / 256, 256, 0, stream>>>(PMT, P(IN_PMASK));
  tri_att(BUF0, PMT, TAE, true);

  // ---- Pair transition (SwiGLU), chunked to bound workspace --------------
  ln(zr, P(TRZ + 0), P(TRZ + 1), BUF2, NSQ, 128);
  for (int ch = 0; ch < 4; ++ch) {
    int m0 = ch * 9216;
    const float* xc = BUF2 + (size_t)m0 * 128;
    float* H1 = BUF0;
    float* H2 = BUF0 + (size_t)9216 * 512;
    gemm(xc, P(TRZ + 2), nullptr, H1, 9216, 128, 512, 2);             // silu(x@w1)
    gemm(xc, P(TRZ + 3), nullptr, H2, 9216, 128, 512, 0);             // x@w2
    ewmul(H1, H1, H2, (size_t)9216 * 512);
    gemm(H1, P(TRZ + 4), nullptr, BUF1, 9216, 512, 128, 0);           // @w3
    ewadd(zr + (size_t)m0 * 128, BUF1, (size_t)9216 * 128);
  }

  // ---- AttentionPairBias on s --------------------------------------------
  ln(sr, P(P_PRE_G), P(P_PRE_B), SN, NRES, 384);
  gemm(SN, P(A_WQ), P(A_BQ), SQ, NRES, 384, 384, 0);
  gemm(SN, P(A_WK), nullptr, SK, NRES, 384, 384, 0);
  gemm(SN, P(A_WV), nullptr, SV, NRES, 384, 384, 0);
  gemm(SN, P(A_WG), nullptr, SG, NRES, 384, 384, 1);
  ln(zr, P(A_ZLN_G), P(A_ZLN_B), BUF2, NSQ, 128);
  gemm(BUF2, P(A_WZ), nullptr, TBUF, NSQ, 128, 16, 0);                // zb [i,j,h]
  s_att_kernel<<<dim3(6, 16), 256, 0, stream>>>(
      SQ, SK, SV, TBUF, P(IN_MASK), SO, 0.20412414523193154f);        // 1/sqrt(24)
  ewmul(SQ, SG, SO, (size_t)NRES * TSC);                              // g*o
  gemm(SQ, P(A_WO), nullptr, SK, NRES, 384, 384, 0);
  ewadd(sr, SK, (size_t)NRES * TSC);

  // ---- Single transition (SwiGLU) ----------------------------------------
  ln(sr, P(TRS + 0), P(TRS + 1), SN, NRES, 384);
  gemm(SN, P(TRS + 2), nullptr, SH1, NRES, 384, 1536, 2);
  gemm(SN, P(TRS + 3), nullptr, SH2, NRES, 384, 1536, 0);
  ewmul(SH1, SH1, SH2, (size_t)NRES * 1536);
  gemm(SH1, P(TRS + 4), nullptr, SO, NRES, 1536, 384, 0);
  ewadd(sr, SO, (size_t)NRES * TSC);
}
